// OrthoLinear_1348619731307
// MI455X (gfx1250) — matrix operand order repaired and checked by reading, not compiled, not run on hardware
//
#include <hip/hip_runtime.h>
#include <stdint.h>

#define IN_DIM   4096
#define OUT_DIM  4096
#define KDIM     IN_DIM
#define BK       32
#define LSTR     40                 // halves per LDS row (80B stride -> conflict-free b128)
#define KT       (KDIM / BK)        // 128 k-tiles
#define BM       128
#define BN       128
#define ABUF_E   (BM * LSTR)        // halves per A buffer
#define BBUF_E   (BN * LSTR)        // halves per B buffer

typedef _Float16 v16h __attribute__((ext_vector_type(16)));
typedef float    v8f  __attribute__((ext_vector_type(8)));
typedef int      v4i  __attribute__((ext_vector_type(4)));

union Frag16 { v16h v; uint4 q[2]; };
union Half16 { _Float16 h[16]; uint4 q[2]; };

#define AS1 __attribute__((address_space(1)))
#define AS3 __attribute__((address_space(3)))

// ---- CDNA5 async copy global->LDS (ASYNCcnt-tracked), 16B per lane ----------
static __device__ __forceinline__ void async_cp16(_Float16* lds, const _Float16* glb) {
    _Float16* g = const_cast<_Float16*>(glb);
#if __has_builtin(__builtin_amdgcn_global_load_async_to_lds_b128)
    __builtin_amdgcn_global_load_async_to_lds_b128(
        (AS1 v4i*)g, (AS3 v4i*)lds, 0, 0);
#else
    asm volatile("global_load_async_to_lds_b128 %0, %1, off"
                 :: "v"((unsigned)(uintptr_t)(AS3 void*)lds), "v"(g)
                 : "memory");
#endif
}

static __device__ __forceinline__ void wait_async0() {
#if __has_builtin(__builtin_amdgcn_s_wait_asynccnt)
    __builtin_amdgcn_s_wait_asynccnt(0);
#else
    asm volatile("s_wait_asynccnt 0" ::: "memory");
#endif
}

// ---------------------------------------------------------------------------
// Kernel 1: dequant int4 base weights -> f16 W_eff (one thread per packed byte)
// ---------------------------------------------------------------------------
__global__ __launch_bounds__(256) void ortho_dequant_kernel(
    const int* __restrict__ base_packed, const float* __restrict__ scales,
    uint32_t* __restrict__ W_u32)
{
    int idx = blockIdx.x * 256 + threadIdx.x;       // over OUT * IN/2
    int o   = idx >> 11;                            // / (IN/2 = 2048)
    int b   = base_packed[idx];
    float s = scales[o];
    float lo = (float)((b & 0xF) - 8) * s;
    float hi = (float)(((b >> 4) & 0xF) - 8) * s;
    _Float16 h2[2] = { (_Float16)lo, (_Float16)hi };
    uint32_t packed;
    __builtin_memcpy(&packed, h2, 4);
    W_u32[idx] = packed;
}

// ---------------------------------------------------------------------------
// Kernel 2: sparse fp16 outliers; unique (row,col) -> race-free RMW add.
// ---------------------------------------------------------------------------
__global__ __launch_bounds__(256) void ortho_scatter_kernel(
    const int* __restrict__ rows, const int* __restrict__ cols,
    const _Float16* __restrict__ vals, const float* __restrict__ alpha,
    _Float16* __restrict__ W, int nnz)
{
    int i = blockIdx.x * 256 + threadIdx.x;
    if (i >= nnz) return;
    int r = rows[i], c = cols[i];
    float add = alpha[0] * (float)vals[i];
    size_t off = (size_t)r * IN_DIM + c;
    W[off] = (_Float16)((float)W[off] + add);
}

// ---------------------------------------------------------------------------
// Kernel 3: convert x f32 -> f16 once (memory-bound; x16 stays hot in L2)
// ---------------------------------------------------------------------------
__global__ __launch_bounds__(256) void ortho_xcvt_kernel(
    const float* __restrict__ x, uint4* __restrict__ x16)
{
    size_t i = (size_t)blockIdx.x * 256 + threadIdx.x;   // per 8 floats
    const float4* p = (const float4*)(x + i * 8);
    float4 f0 = p[0], f1 = p[1];
    Half16 h;
    h.h[0] = (_Float16)f0.x; h.h[1] = (_Float16)f0.y;
    h.h[2] = (_Float16)f0.z; h.h[3] = (_Float16)f0.w;
    h.h[4] = (_Float16)f1.x; h.h[5] = (_Float16)f1.y;
    h.h[6] = (_Float16)f1.z; h.h[7] = (_Float16)f1.w;
    x16[i] = h.q[0];
}

// ---------------------------------------------------------------------------
// Kernel 4 (primary): 128x128 block, 8 wave32s in 4x2 grid, 32x64 per wave.
// Parity-free double buffering (integer offsets, unroll disabled) so the
// accumulators keep stable registers -> no v_mov/v_nop churn around WMMAs.
// LDS filled by GLOBAL_LOAD_ASYNC_TO_LDS_B128, ASYNCcnt-synchronized.
// ---------------------------------------------------------------------------
__global__ __launch_bounds__(256) void ortho_wmma_gemm_async_kernel(
    const _Float16* __restrict__ x16, const _Float16* __restrict__ W,
    float* __restrict__ out)
{
    __shared__ __align__(16) _Float16 Abuf[2 * ABUF_E];
    __shared__ __align__(16) _Float16 Bbuf[2 * BBUF_E];

    const int t    = threadIdx.x;
    const int lane = t & 31;
    const int wv   = t >> 5;
    const int wm   = wv & 3;          // 4 waves along M, 32 rows each
    const int wn   = wv >> 2;         // 2 waves along N, 64 cols each
    const int hs   = lane >> 4;
    const int lrow = lane & 15;

    const int m0 = blockIdx.y * BM;
    const int n0 = blockIdx.x * BN;

    // each thread DMAs 32B of A and 32B of B per k-step
    const int lr = t >> 1;            // 0..127
    const int lc = (t & 1) * 16;      // 0 / 16

    const _Float16* ag = x16 + (size_t)(m0 + lr) * KDIM + lc;
    const _Float16* bg = W   + (size_t)(n0 + lr) * KDIM + lc;

    _Float16* a_st = &Abuf[lr * LSTR + lc];      // store base, buffer 0
    _Float16* b_st = &Bbuf[lr * LSTR + lc];

    // compute-side base addresses (buffer 0)
    const _Float16* a_ld = &Abuf[(wm * 32 + lrow) * LSTR + hs * 8];
    const _Float16* b_ld = &Bbuf[(wn * 64 + lrow) * LSTR + hs * 8];

    // prologue: async-load k-tile 0 into buffer 0
    async_cp16(a_st,     ag);
    async_cp16(a_st + 8, ag + 8);
    async_cp16(b_st,     bg);
    async_cp16(b_st + 8, bg + 8);

    v8f acc[2][4];
    #pragma unroll
    for (int mt = 0; mt < 2; ++mt)
        #pragma unroll
        for (int nt = 0; nt < 4; ++nt) {
            v8f z = {0.f,0.f,0.f,0.f,0.f,0.f,0.f,0.f};
            acc[mt][nt] = z;
        }

    #pragma clang loop unroll(disable)
    for (int kt = 0; kt < KT; ++kt) {
        wait_async0();                // this wave's copies of tile kt landed
        __syncthreads();              // ... and everyone else's

        const int aoff = (kt & 1) * ABUF_E;      // current buffers
        const int boff = (kt & 1) * BBUF_E;

        // WAR-safe: all waves' reads of the other buffer completed before
        // this barrier. Kick off DMA for tile kt+1 into it.
        if (kt + 1 < KT) {
            const _Float16* agk = ag + (size_t)(kt + 1) * BK;
            const _Float16* bgk = bg + (size_t)(kt + 1) * BK;
            _Float16* al = a_st + (ABUF_E - aoff);
            _Float16* bl = b_st + (BBUF_E - boff);
            async_cp16(al,     agk);
            async_cp16(al + 8, agk + 8);
            async_cp16(bl,     bgk);
            async_cp16(bl + 8, bgk + 8);
        }

        // fragment loads + 8 WMMAs on current tile
        Frag16 a[2], b[4];
        #pragma unroll
        for (int mt = 0; mt < 2; ++mt) {
            const _Float16* p = a_ld + aoff + mt * (16 * LSTR);
            a[mt].q[0] = *(const uint4*)p;
            a[mt].q[1] = *(const uint4*)(p + 16);
        }
        #pragma unroll
        for (int nt = 0; nt < 4; ++nt) {
            const _Float16* p = b_ld + boff + nt * (16 * LSTR);
            b[nt].q[0] = *(const uint4*)p;
            b[nt].q[1] = *(const uint4*)(p + 16);
        }
        #pragma unroll
        for (int mt = 0; mt < 2; ++mt)
            #pragma unroll
            for (int nt = 0; nt < 4; ++nt)
                acc[mt][nt] = __builtin_amdgcn_wmma_f32_16x16x32_f16(
                    false, a[mt].v, false, b[nt].v,
                    (short)0, acc[mt][nt], false, false);
    }

    // epilogue: lanes 0-15 -> M=v,N=lane; lanes 16-31 -> M=v+8,N=lane-16
    #pragma unroll
    for (int mt = 0; mt < 2; ++mt) {
        #pragma unroll
        for (int nt = 0; nt < 4; ++nt) {
            const int gm = m0 + wm*32 + mt*16 + hs*8;
            const int gn = n0 + wn*64 + nt*16 + lrow;
            float* op = out + (size_t)gm * OUT_DIM + gn;
            #pragma unroll
            for (int v = 0; v < 8; ++v)
                __builtin_nontemporal_store(acc[mt][nt][v], op + (size_t)v * OUT_DIM);
        }
    }
}

// ---------------------------------------------------------------------------
// Fallback GEMM (round-1 style): used only if workspace can't hold x16.
// ---------------------------------------------------------------------------
__global__ __launch_bounds__(256) void ortho_wmma_gemm_kernel(
    const float* __restrict__ x, const _Float16* __restrict__ W,
    float* __restrict__ out)
{
    __shared__ __align__(16) _Float16 Abuf[2][BM * LSTR];
    __shared__ __align__(16) _Float16 Bbuf[2][BN * LSTR];

    const int t    = threadIdx.x;
    const int lane = t & 31;
    const int wv   = t >> 5;
    const int wm   = wv & 3;
    const int wn   = wv >> 2;
    const int hs   = lane >> 4;
    const int lrow = lane & 15;

    const int m0 = blockIdx.y * BM;
    const int n0 = blockIdx.x * BN;

    const int lr = t >> 1;
    const int lc = (t & 1) * 16;

    const float*    xg = x + (size_t)(m0 + lr) * KDIM + lc;
    const _Float16* wg = W + (size_t)(n0 + lr) * KDIM + lc;

    float4 af[4];
    uint4  bf[2];
    {
        const float4* ap = (const float4*)xg;
        af[0] = ap[0]; af[1] = ap[1]; af[2] = ap[2]; af[3] = ap[3];
        const uint4* bp = (const uint4*)wg;
        bf[0] = bp[0]; bf[1] = bp[1];
    }
    {
        Half16 ac;
        #pragma unroll
        for (int q = 0; q < 4; ++q) {
            ac.h[q*4+0] = (_Float16)af[q].x; ac.h[q*4+1] = (_Float16)af[q].y;
            ac.h[q*4+2] = (_Float16)af[q].z; ac.h[q*4+3] = (_Float16)af[q].w;
        }
        _Float16* as = &Abuf[0][lr * LSTR + lc];
        *(uint4*)as = ac.q[0]; *(uint4*)(as + 8) = ac.q[1];
        _Float16* bs = &Bbuf[0][lr * LSTR + lc];
        *(uint4*)bs = bf[0];   *(uint4*)(bs + 8) = bf[1];
    }

    v8f acc[2][4];
    #pragma unroll
    for (int mt = 0; mt < 2; ++mt)
        #pragma unroll
        for (int nt = 0; nt < 4; ++nt) {
            v8f z = {0.f,0.f,0.f,0.f,0.f,0.f,0.f,0.f};
            acc[mt][nt] = z;
        }

    const int koff = hs * 8;

    for (int kt = 0; kt < KT; ++kt) {
        __syncthreads();
        const int cur = kt & 1;
        if (kt + 1 < KT) {
            const float4* ap = (const float4*)(xg + (size_t)(kt + 1) * BK);
            af[0] = ap[0]; af[1] = ap[1]; af[2] = ap[2]; af[3] = ap[3];
            const uint4* bp = (const uint4*)(wg + (size_t)(kt + 1) * BK);
            bf[0] = bp[0]; bf[1] = bp[1];
        }
        Frag16 a[2], b[4];
        const _Float16* Ab = Abuf[cur];
        const _Float16* Bb = Bbuf[cur];
        #pragma unroll
        for (int mt = 0; mt < 2; ++mt) {
            const _Float16* p = Ab + (wm*32 + mt*16 + lrow) * LSTR + koff;
            a[mt].q[0] = *(const uint4*)p;
            a[mt].q[1] = *(const uint4*)(p + 16);
        }
        #pragma unroll
        for (int nt = 0; nt < 4; ++nt) {
            const _Float16* p = Bb + (wn*64 + nt*16 + lrow) * LSTR + hs*8;
            b[nt].q[0] = *(const uint4*)p;
            b[nt].q[1] = *(const uint4*)(p + 16);
        }
        #pragma unroll
        for (int mt = 0; mt < 2; ++mt)
            #pragma unroll
            for (int nt = 0; nt < 4; ++nt)
                acc[mt][nt] = __builtin_amdgcn_wmma_f32_16x16x32_f16(
                    false, a[mt].v, false, b[nt].v,
                    (short)0, acc[mt][nt], false, false);
        __syncthreads();
        if (kt + 1 < KT) {
            const int nxt = cur ^ 1;
            Half16 ac;
            #pragma unroll
            for (int q = 0; q < 4; ++q) {
                ac.h[q*4+0] = (_Float16)af[q].x; ac.h[q*4+1] = (_Float16)af[q].y;
                ac.h[q*4+2] = (_Float16)af[q].z; ac.h[q*4+3] = (_Float16)af[q].w;
            }
            _Float16* as = &Abuf[nxt][lr * LSTR + lc];
            *(uint4*)as = ac.q[0]; *(uint4*)(as + 8) = ac.q[1];
            _Float16* bs = &Bbuf[nxt][lr * LSTR + lc];
            *(uint4*)bs = bf[0];   *(uint4*)(bs + 8) = bf[1];
        }
    }

    #pragma unroll
    for (int mt = 0; mt < 2; ++mt) {
        #pragma unroll
        for (int nt = 0; nt < 4; ++nt) {
            const int gm = m0 + wm*32 + mt*16 + hs*8;
            const int gn = n0 + wn*64 + nt*16 + lrow;
            float* op = out + (size_t)gm * OUT_DIM + gn;
            #pragma unroll
            for (int v = 0; v < 8; ++v)
                __builtin_nontemporal_store(acc[mt][nt][v], op + (size_t)v * OUT_DIM);
        }
    }
}

// ---------------------------------------------------------------------------
extern "C" void kernel_launch(void* const* d_in, const int* in_sizes, int n_in,
                              void* d_out, int out_size, void* d_ws, size_t ws_size,
                              hipStream_t stream) {
    (void)n_in; (void)out_size;
    const float*    x           = (const float*)d_in[0];
    const int*      base_packed = (const int*)d_in[1];
    const float*    scales      = (const float*)d_in[2];
    const _Float16* ortho_vals  = (const _Float16*)d_in[3];
    const int*      ortho_rows  = (const int*)d_in[4];
    const int*      ortho_cols  = (const int*)d_in[5];
    const float*    alpha       = (const float*)d_in[6];
    float*          out         = (float*)d_out;
    _Float16*       W           = (_Float16*)d_ws;        // 32 MiB

    const int packed_elems = in_sizes[1];                 // 8388608
    ortho_dequant_kernel<<<packed_elems / 256, 256, 0, stream>>>(
        base_packed, scales, (uint32_t*)W);

    const int nnz = in_sizes[3];                          // 839680
    ortho_scatter_kernel<<<(nnz + 255) / 256, 256, 0, stream>>>(
        ortho_rows, ortho_cols, ortho_vals, alpha, W, nnz);

    const int M = in_sizes[0] / IN_DIM;                   // 8192
    const size_t wbytes = (size_t)OUT_DIM * IN_DIM * sizeof(_Float16);
    const size_t xbytes = (size_t)M * KDIM * sizeof(_Float16);

    if (ws_size >= wbytes + xbytes) {
        _Float16* x16 = (_Float16*)((char*)d_ws + wbytes);
        const int cvt_blocks = (int)(((size_t)M * KDIM / 8) / 256);
        ortho_xcvt_kernel<<<cvt_blocks, 256, 0, stream>>>(x, (uint4*)x16);
        dim3 grid(OUT_DIM / BN, M / BM);                  // 32 x 64
        ortho_wmma_gemm_async_kernel<<<grid, 256, 0, stream>>>(x16, W, out);
    } else {
        dim3 grid(OUT_DIM / BN, M / BM);                  // 32 x 64
        ortho_wmma_gemm_kernel<<<grid, 256, 0, stream>>>(x, W, out);
    }
}
